// SDT_59923383713865
// MI455X (gfx1250) — matrix operand order repaired and checked by reading, not compiled, not run on hardware
//
#include <hip/hip_runtime.h>
#include <cstdint>

typedef __attribute__((ext_vector_type(16))) __bf16 v16bf;
typedef __attribute__((ext_vector_type(8)))  float  v8f;

#define BATCH     16384
#define IN_DIM    4096
#define OUT_DIM   1000
#define NUM_INNER 255
#define ALLP      511

__device__ __forceinline__ unsigned short f2bf(float f) {
  union { __bf16 h; unsigned short u; } c;
  c.h = (__bf16)f;                     // native fptrunc, RNE
  return c.u;
}
__device__ __forceinline__ unsigned int pk2bf(float a, float b) {
  return (unsigned)f2bf(a) | ((unsigned)f2bf(b) << 16);
}
// Low 32 bits of a generic pointer into __shared__ = LDS byte offset (LDS
// aperture maps addr[31:0] to the wave's LDS allocation).
__device__ __forceinline__ unsigned lds_u32(const void* p) {
  return (unsigned)(uintptr_t)p;
}
__device__ __forceinline__ void async_ld_b128(unsigned lds_dst, unsigned goff,
                                              const void* sbase) {
  asm volatile("global_load_async_to_lds_b128 %0, %1, %2"
               :: "v"(lds_dst), "v"(goff), "s"(sbase) : "memory");
}
__device__ __forceinline__ void wait_async0() {
  asm volatile("s_wait_asynccnt 0x0" ::: "memory");
}

// ---------------------------------------------------------------------------
// Tiled WMMA GEMM:  C[M x N] = act( A[M x K] * Bbf[N x K]^T + bias )
//  - B is pre-converted bf16 (padded rows), staged via async global->LDS DMA.
//  - A: ABF16 ? pre-packed bf16 (async DMA) : fp32 converted inline.
// Block tile 64x128, 8 waves (4 row x 2 col groups), wave = 16x64, K-step 32.
// LDS row stride 40 ushorts (80B) keeps every 16B chunk aligned.
// ---------------------------------------------------------------------------
template<bool SIG, bool ABF16>
__global__ __launch_bounds__(256)
void k_gemm(const void* __restrict__ Aptr, const unsigned short* __restrict__ Bbf,
            const float* __restrict__ biasP, float* __restrict__ C,
            int K, int N_real, long lda, long ldb, long ldc, long biasStride)
{
  __shared__ unsigned short sA[64 * 40];    // [row][k]
  __shared__ unsigned short sB[128 * 40];   // [col][k]

  const int tid  = threadIdx.x;
  const int lane = tid & 31;
  const int wid  = tid >> 5;
  const int wr   = wid & 3;
  const int wc   = wid >> 2;
  const int l16  = lane & 15;
  const bool hi  = lane >= 16;
  const long row0 = (long)blockIdx.x * 64;
  const long col0 = (long)blockIdx.y * 128;

  // --- async-copy descriptors: each 16B chunk = (row, 8 bf16) of a tile
  const int crow = tid >> 2;                 // chunk row 0..63
  const int ck   = (tid & 3) * 8;            // chunk k-origin (elements)
  const unsigned short* baseB = Bbf + col0 * ldb;
  const unsigned ldsB0 = lds_u32(&sB[crow * 40 + ck]);
  const unsigned ldsB1 = lds_u32(&sB[(crow + 64) * 40 + ck]);
  unsigned gB0 = (unsigned)(crow * ldb * 2 + (tid & 3) * 16);
  unsigned gB1 = gB0 + (unsigned)(64 * ldb * 2);

  const unsigned short* baseA16 = (const unsigned short*)Aptr + row0 * lda;
  const unsigned ldsA = lds_u32(&sA[crow * 40 + ck]);
  unsigned gA = (unsigned)(crow * lda * 2 + (tid & 3) * 16);

  v8f zero = {0,0,0,0,0,0,0,0};
  v8f acc[4] = {zero, zero, zero, zero};

  for (int k0 = 0; k0 < K; k0 += 32) {
    if constexpr (ABF16) {
      async_ld_b128(ldsA, gA, baseA16);
      gA += 64;
    } else {
      // stage A: 64 rows x 32 k fp32 -> bf16 (X streamed once; prefetch ahead)
      const float* A = (const float*)Aptr;
      #pragma unroll
      for (int it = 0; it < 2; ++it) {
        int lin = tid + it * 256;
        int r   = lin >> 3;
        int kq  = (lin & 7) << 2;
        const float* ap = A + (row0 + r) * lda + k0 + kq;
        if (it == 0) __builtin_prefetch(ap + 128, 0, 0);
        const float4 a = *(const float4*)ap;
        *(uint2*)&sA[r * 40 + kq] = make_uint2(pk2bf(a.x, a.y), pk2bf(a.z, a.w));
      }
    }
    // stage B: 128 cols x 32 k bf16 via async DMA (2 chunks/thread)
    async_ld_b128(ldsB0, gB0, baseB);
    async_ld_b128(ldsB1, gB1, baseB);
    gB0 += 64; gB1 += 64;

    wait_async0();
    __syncthreads();

    union { v16bf v; uint4 q[2]; } af, bfv;
    const unsigned short* aRow = &sA[(wr * 16 + l16) * 40];
    const int ao = hi ? 8 : 0;
    af.q[0] = *(const uint4*)(aRow + ao);
    af.q[1] = *(const uint4*)(aRow + ao + 16);

    #pragma unroll
    for (int t = 0; t < 4; ++t) {
      const unsigned short* bCol = &sB[(wc * 64 + t * 16 + l16) * 40];
      const int bo = hi ? 8 : 0;
      bfv.q[0] = *(const uint4*)(bCol + bo);
      bfv.q[1] = *(const uint4*)(bCol + bo + 16);
      acc[t] = __builtin_amdgcn_wmma_f32_16x16x32_bf16(
          false, af.v, false, bfv.v, (short)0, acc[t], false, false);
    }
    __syncthreads();
  }

  #pragma unroll
  for (int t = 0; t < 4; ++t) {
    long gc = col0 + wc * 64 + t * 16 + l16;
    float bias = 0.0f;
    if (SIG) bias = (gc < N_real) ? biasP[gc * biasStride] : 0.0f;
    long grow = row0 + wr * 16 + (hi ? 8 : 0);
    #pragma unroll
    for (int r = 0; r < 8; ++r) {
      float v = acc[t][r] + bias;
      if (SIG) v = 1.0f / (1.0f + __expf(-v));
      if (gc < N_real) C[(grow + r) * ldc + gc] = v;
    }
  }
}

// ---------------------------------------------------------------------------
// Pre-convert weights: W[rows_real x (2^lc)] fp32 (stride ldw, col offset)
// -> bf16 padded [rows_pad x 2^lc], zero-filled pad rows. 8 elems/thread.
// ---------------------------------------------------------------------------
__global__ __launch_bounds__(256)
void k_convW(const float* __restrict__ W, unsigned short* __restrict__ Wb,
             int rows_real, int lc, long ldw, long col_off)
{
  long idx = (((long)blockIdx.x * 256) + threadIdx.x) * 8;
  long r = idx >> lc;
  long c = idx & ((1L << lc) - 1);
  unsigned int o0 = 0, o1 = 0, o2 = 0, o3 = 0;
  if (r < rows_real) {
    const float* p = W + r * ldw + col_off + c;
    o0 = pk2bf(p[0], p[1]); o1 = pk2bf(p[2], p[3]);
    o2 = pk2bf(p[4], p[5]); o3 = pk2bf(p[6], p[7]);
  }
  *(uint4*)&Wb[idx] = make_uint4(o0, o1, o2, o3);
}

// ---------------------------------------------------------------------------
// Tree pass depth d: mu_d -> children, alpha sums; d==7 also emits final mu
// (fp32 output + packed bf16 for the logits GEMM).
// ---------------------------------------------------------------------------
__global__ __launch_bounds__(256)
void k_tree_pass(const float* __restrict__ P, float* __restrict__ allp,
                 float* __restrict__ muOut, unsigned int* __restrict__ muBf,
                 float* __restrict__ S0, float* __restrict__ S1, int d)
{
  __shared__ float s0[128], s1[128];
  const int n   = 1 << d;
  const int tid = threadIdx.x;
  if (tid < n) { s0[tid] = 0.0f; s1[tid] = 0.0f; }
  __syncthreads();

  long idx = (long)blockIdx.x * 256 + tid;
  int  i   = (int)(idx & (long)(n - 1));
  long b   = idx >> d;
  const long rowA = b * ALLP;

  float mu = (d == 0) ? 1.0f : allp[rowA + (n - 1) + i];
  float p  = P[b * 256 + (n - 1) + i];
  atomicAdd(&s1[i], p * mu);
  atomicAdd(&s0[i], mu);

  float cl = mu * (1.0f - p);
  float cr = mu * p;
  long cbase = rowA + (2 * n - 1) + 2 * i;
  allp[cbase]     = cl;
  allp[cbase + 1] = cr;
  if (d == 0) allp[rowA] = 1.0f;
  if (d == 7) {
    long mb = b * 256 + 2 * i;
    muOut[mb]     = cl;
    muOut[mb + 1] = cr;
    muBf[b * 128 + i] = pk2bf(cl, cr);
  }
  __syncthreads();
  if (tid < n) {
    atomicAdd(&S0[(n - 1) + tid], s0[tid]);
    atomicAdd(&S1[(n - 1) + tid], s1[tid]);
  }
}

// ---------------------------------------------------------------------------
// Missingness penalty (factored):  sum_n decay(n) * dot(colsum(M), |W[n,1:]|)
// ---------------------------------------------------------------------------
__global__ __launch_bounds__(256)
void k_colsum_partial(const float* __restrict__ Mm, float* __restrict__ part)
{
  int  c  = blockIdx.x * 256 + threadIdx.x;
  long r0 = (long)blockIdx.y * 256;
  float s = 0.0f;
  for (int r = 0; r < 256; ++r)
    s += Mm[(r0 + r) * (long)IN_DIM + c];
  part[(long)blockIdx.y * IN_DIM + c] = s;
}

__global__ __launch_bounds__(256)
void k_colsum_reduce(const float* __restrict__ part, float* __restrict__ colM)
{
  int c = blockIdx.x * 256 + threadIdx.x;
  float s = 0.0f;
  for (int j = 0; j < 64; ++j) s += part[(long)j * IN_DIM + c];
  colM[c] = s;
}

__global__ __launch_bounds__(256)
void k_missing(const float* __restrict__ colM, const float* __restrict__ Wi,
               float* __restrict__ pMiss)
{
  __shared__ float red[256];
  int nn = blockIdx.x;
  const float* wrow = Wi + (long)nn * 4097 + 1;
  float s = 0.0f;
  for (int dcol = threadIdx.x; dcol < IN_DIM; dcol += 256)
    s += colM[dcol] * fabsf(wrow[dcol]);
  red[threadIdx.x] = s; __syncthreads();
  for (int off = 128; off > 0; off >>= 1) {
    if (threadIdx.x < off) red[threadIdx.x] += red[threadIdx.x + off];
    __syncthreads();
  }
  if (threadIdx.x == 0) {
    int depth = 31 - __clz(nn + 1);
    atomicAdd(pMiss, red[0] / (float)(1 << depth));
  }
}

__global__ __launch_bounds__(256)
void k_split(const float* __restrict__ S0, const float* __restrict__ S1,
             float* __restrict__ pSplit)
{
  __shared__ float red[256];
  float s = 0.0f;
  for (int nn = threadIdx.x; nn < NUM_INNER; nn += 256) {
    int depth = 31 - __clz(nn + 1);
    float a = S1[nn] / S0[nn];
    float t = __logf(a) + __logf(1.0f - a);
    if (!__builtin_isfinite(t)) t = 0.0f;
    s += -0.5f * t / (float)(1 << depth);
  }
  red[threadIdx.x] = s; __syncthreads();
  for (int off = 128; off > 0; off >>= 1) {
    if (threadIdx.x < off) red[threadIdx.x] += red[threadIdx.x + off];
    __syncthreads();
  }
  if (threadIdx.x == 0) *pSplit = red[0];
}

__global__ void k_init(float* S0, float* S1, float* pSplit, float* pMiss)
{
  int t = threadIdx.x;
  if (t < 256) { S0[t] = 0.0f; S1[t] = 0.0f; }
  if (t == 0)  { *pSplit = 0.0f; *pMiss = 0.0f; }
}

// ---------------------------------------------------------------------------
extern "C" void kernel_launch(void* const* d_in, const int* in_sizes, int n_in,
                              void* d_out, int out_size, void* d_ws, size_t ws_size,
                              hipStream_t stream)
{
  const float* X  = (const float*)d_in[0];   // (16384, 4096)
  const float* Mm = (const float*)d_in[1];   // (16384, 4096)
  const float* Wi = (const float*)d_in[2];   // (255, 4097)
  const float* Wl = (const float*)d_in[3];   // (1000, 256)
  float* out = (float*)d_out;

  float* logits = out;                                  // 16384*1000
  float* muOut  = logits + (long)BATCH * OUT_DIM;       // 16384*256
  float* allp   = muOut  + (long)BATCH * 256;           // 16384*511
  float* pSplit = allp   + (long)BATCH * ALLP;          // scalar
  float* pMiss  = pSplit + 1;                           // scalar

  char* w = (char*)d_ws;
  float* P    = (float*)w;                 w += (size_t)BATCH * 256 * 4;
  float* part = (float*)w;                 w += 64L * IN_DIM * 4;
  float* colM = (float*)w;                 w += IN_DIM * 4;
  float* S0   = (float*)w;                 w += 256 * 4;
  float* S1   = (float*)w;                 w += 256 * 4;
  unsigned short* muBf = (unsigned short*)w; w += (size_t)BATCH * 256 * 2;
  unsigned short* Wbf  = (unsigned short*)w; w += 256L * IN_DIM * 2;   // 256x4096
  unsigned short* Wlbf = (unsigned short*)w; w += 1024L * 256 * 2;     // 1024x256

  k_init<<<1, 256, 0, stream>>>(S0, S1, pSplit, pMiss);

  // one-shot weight bf16 pre-conversion (padded, zero-filled tails)
  k_convW<<<512, 256, 0, stream>>>(Wi, Wbf, NUM_INNER, 12, 4097L, 1L);
  k_convW<<<128, 256, 0, stream>>>(Wl, Wlbf, OUT_DIM, 8, 256L, 0L);

  // missingness penalty (factored rank-1 form)
  k_colsum_partial<<<dim3(IN_DIM / 256, 64), 256, 0, stream>>>(Mm, part);
  k_colsum_reduce<<<IN_DIM / 256, 256, 0, stream>>>(part, colM);
  k_missing<<<NUM_INNER, 256, 0, stream>>>(colM, Wi, pMiss);

  // P = sigmoid(X @ Wi[:,1:]^T + Wi[:,0])  (M=16384, N=255 pad 256, K=4096)
  k_gemm<true, false><<<dim3(BATCH / 64, 2), 256, 0, stream>>>(
      X, Wbf, Wi, P, IN_DIM, NUM_INNER, (long)IN_DIM, (long)IN_DIM, 256L, 4097L);

  // tree recursion: 8 streaming passes
  for (int d = 0; d < 8; ++d) {
    int blocks = (BATCH << d) / 256;
    k_tree_pass<<<blocks, 256, 0, stream>>>(P, allp, muOut, (unsigned int*)muBf,
                                            S0, S1, d);
  }
  k_split<<<1, 256, 0, stream>>>(S0, S1, pSplit);

  // logits = mu @ Wl^T  (M=16384, N=1000 pad 1024, K=256), all-bf16 async path
  k_gemm<false, true><<<dim3(BATCH / 64, 8), 256, 0, stream>>>(
      muBf, Wlbf, nullptr, logits, 256, OUT_DIM, 256L, 256L, (long)OUT_DIM, 0L);
}
